// Spin_14224931684722
// MI455X (gfx1250) — matrix operand order repaired and checked by reading, not compiled, not run on hardware
//
#include <hip/hip_runtime.h>
#include <hip/hip_bf16.h>
#include <math.h>

// ---------------- problem constants ----------------
#define TWO_B   16384      // rows of x / z
#define B_HALF  8192       // rows per parity (z1 / z2), also Sinkhorn "B"
#define INP     1024       // input dim
#define CDIM    256        // codebook dim
#define K_CB    2048       // codebook size, Sinkhorn "K"
#define INV_EPS 50.0f      // 1/epsilon
#define INV_T   10.0f      // 1/temperature

typedef __attribute__((ext_vector_type(16))) _Float16     v16h;
typedef __attribute__((ext_vector_type(8)))  _Float16     v8h;
typedef __attribute__((ext_vector_type(8)))  float        v8f;
typedef __attribute__((ext_vector_type(4)))  unsigned int v4u;
typedef __attribute__((ext_vector_type(8)))  int          v8i;
typedef __attribute__((ext_vector_type(4)))  int          v4i;

// ---------------- prep: convert head_w to f16 ----------------
__global__ void k_cvt_headw(const float* __restrict__ hw, _Float16* __restrict__ out) {
    int i = blockIdx.x * 256 + threadIdx.x;   // exactly CDIM*INP threads launched
    out[i] = (_Float16)hw[i];
}

// ---------------- prep: wn = cb_g * cb_v / ||cb_v||  (f16) ----------------
__global__ void __launch_bounds__(256) k_prep_wn(const float* __restrict__ cb_v,
                                                 const float* __restrict__ cb_g,
                                                 _Float16* __restrict__ wn) {
    __shared__ float red[256];
    int r = blockIdx.x, t = threadIdx.x;
    float v = cb_v[r * CDIM + t];
    red[t] = v * v;
    __syncthreads();
    for (int s = 128; s > 0; s >>= 1) { if (t < s) red[t] += red[t + s]; __syncthreads(); }
    float scale = cb_g[r] * rsqrtf(fmaxf(red[0], 1e-24f));
    wn[r * CDIM + t] = (_Float16)(v * scale);
}

// ---------------- GEMM1: h = rownorm(x @ head_w^T + b) -> f16 ----------------
// M-tile = 16 rows/block. Whole 16x1024 A tile staged to LDS (32 KB f16) once;
// K loop is then barrier-free: ds_load_b128 + global_load_b128 + v_wmma.
__global__ void __launch_bounds__(256) k_head_gemm(const float* __restrict__ x,
                                                   const _Float16* __restrict__ Wh,
                                                   const float* __restrict__ bias,
                                                   _Float16* __restrict__ h16) {
    __shared__ __align__(16) _Float16 As[16 * INP];   // 32 KB
    __shared__ float rowsq[16];

    const int tid   = threadIdx.x;
    const int wave  = tid >> 5;
    const int lane  = tid & 31;
    const int m16   = lane & 15;       // N within tile / M for A-frag
    const int khalf = lane >> 4;       // which K half this lane holds
    const int row0  = blockIdx.x * 16;

    const int n0 = wave * 32 + m16;        // global column of acc0
    const int n1 = n0 + 16;                // global column of acc1

    // stage the full contiguous 16x1024 f32 tile as f16 (float2 per thread per step)
    {
        const float2* xp = (const float2*)(x + (size_t)row0 * INP);
        for (int j = 0; j < 32; ++j) {
            int e2 = j * 256 + tid;          // float2 index
            float2 v = xp[e2];
            As[e2 * 2]     = (_Float16)v.x;
            As[e2 * 2 + 1] = (_Float16)v.y;
        }
    }
    __syncthreads();

    v8f acc0 = {}; v8f acc1 = {};

    for (int kk = 0; kk < INP; kk += 32) {
        // A fragment, contiguous K order matching B: lane<16 -> K 0..15, lane>=16 -> K 16..31
        v8h alo = *(const v8h*)&As[m16 * INP + kk + 16 * khalf];
        v8h ahi = *(const v8h*)&As[m16 * INP + kk + 16 * khalf + 8];
        v16h a  = __builtin_shufflevector(alo, ahi, 0,1,2,3,4,5,6,7,8,9,10,11,12,13,14,15);

        // B fragments: lane holds N = lane%16, 16 contiguous K halves (khalf selects K 0-15 / 16-31)
        v16h b0 = *(const v16h*)&Wh[(size_t)n0 * INP + kk + 16 * khalf];
        v16h b1 = *(const v16h*)&Wh[(size_t)n1 * INP + kk + 16 * khalf];

        if (kk + 32 < INP) {
            __builtin_prefetch(&Wh[(size_t)n0 * INP + kk + 32], 0, 0);   // global_prefetch_b8
            __builtin_prefetch(&Wh[(size_t)n1 * INP + kk + 32], 0, 0);
        }

        acc0 = __builtin_amdgcn_wmma_f32_16x16x32_f16(false, a, false, b0, (short)0, acc0, false, false);
        acc1 = __builtin_amdgcn_wmma_f32_16x16x32_f16(false, a, false, b1, (short)0, acc1, false, false);
    }

    // bias (C/D layout: lane -> N = lane%16, VGPR r -> M = r + 8*khalf)
    float bn0 = bias[n0], bn1 = bias[n1];
    for (int r = 0; r < 8; ++r) { acc0[r] += bn0; acc1[r] += bn1; }

    if (tid < 16) rowsq[tid] = 0.f;
    __syncthreads();
    const int rowb = 8 * khalf;
    for (int r = 0; r < 8; ++r)
        atomicAdd(&rowsq[rowb + r], acc0[r] * acc0[r] + acc1[r] * acc1[r]);   // ds_add_f32
    __syncthreads();

    for (int r = 0; r < 8; ++r) {
        int m = rowb + r;
        float inv = 1.0f / fmaxf(sqrtf(rowsq[m]), 1e-12f);
        h16[(size_t)(row0 + m) * CDIM + n0] = (_Float16)(acc0[r] * inv);
        h16[(size_t)(row0 + m) * CDIM + n1] = (_Float16)(acc1[r] * inv);
    }
}

// ---------------- GEMM2: z = h @ wn^T  (f32 out) ----------------
// grid (TWO_B/16, K_CB/128). A tile (16x256 f16 = 8 KB, a contiguous 2-D tile of h16)
// is DMA'd into LDS by the Tensor Data Mover: wave 0 issues tensor_load_to_lds with a
// 2-D D# (data_size=2B, tile 256x16, tensor_dim0_stride=256), waits on TENSORcnt,
// then the block barriers before the WMMA K loop.
__global__ void __launch_bounds__(256) k_z_gemm(const _Float16* __restrict__ h16,
                                                const _Float16* __restrict__ wn,
                                                float* __restrict__ z) {
    __shared__ __align__(16) _Float16 As[16 * CDIM];  // 8 KB

    const int tid   = threadIdx.x;
    const int wave  = tid >> 5;
    const int lane  = tid & 31;
    const int m16   = lane & 15;
    const int khalf = lane >> 4;
    const int row0  = blockIdx.x * 16;
    const int n     = blockIdx.y * 128 + wave * 16 + m16;

    if (wave == 0) {
        unsigned long long ga = (unsigned long long)(h16 + (size_t)row0 * CDIM);
        unsigned int lds = (unsigned int)(unsigned long long)(&As[0]);  // flat LDS addr: low 32 bits = LDS offset

        // D# group 0: count=1 | lds_addr | global_addr[56:0] | type=2 (bits 127:126)
        v4u g0 = { 1u, lds, (unsigned int)ga,
                   (unsigned int)((ga >> 32) & 0x01FFFFFFull) | 0x80000000u };
        // D# group 1: data_size=2B (code 1 @ bits17:16); tensor_dim0=256; tensor_dim1=TWO_B;
        // tile_dim0=256; tile_dim1=16; tile_dim2=0; tensor_dim0_stride=256; tensor_dim1_stride=0
        v8i g1 = { (int)0x00010000,        // workgroup_mask=0, data_size=1(2B)
                   (int)(CDIM << 16),      // tensor_dim0[15:0] @ bits 63:48
                   (int)(TWO_B << 16),     // tensor_dim0[31:16]=0, tensor_dim1[15:0] @ bits 95:80
                   (int)(CDIM << 16),      // tensor_dim1[31:16]=0, tile_dim0 @ bits 127:112
                   (int)16,                // tile_dim1 @ bits 143:128, tile_dim2=0
                   (int)CDIM,              // tensor_dim0_stride[31:0]
                   0, 0 };                 // stride hi, tensor_dim1_stride=0
        v4i g2 = { 0, 0, 0, 0 };
        v4i g3 = { 0, 0, 0, 0 };
        v8i g4 = { 0, 0, 0, 0, 0, 0, 0, 0 };   // extra group of the 6-arg clang-23 builtin (unused, 2-D tensor)
        __builtin_amdgcn_tensor_load_to_lds(g0, g1, g2, g3, g4, 0);
        __builtin_amdgcn_s_wait_tensorcnt(0);
    }
    __syncthreads();

    v8f acc = {};
    for (int kk = 0; kk < CDIM; kk += 32) {
        v8h alo = *(const v8h*)&As[m16 * CDIM + kk + 16 * khalf];
        v8h ahi = *(const v8h*)&As[m16 * CDIM + kk + 16 * khalf + 8];
        v16h a  = __builtin_shufflevector(alo, ahi, 0,1,2,3,4,5,6,7,8,9,10,11,12,13,14,15);
        v16h b  = *(const v16h*)&wn[(size_t)n * CDIM + kk + 16 * khalf];
        if (kk + 32 < CDIM)
            __builtin_prefetch(&wn[(size_t)n * CDIM + kk + 32], 0, 0);
        acc = __builtin_amdgcn_wmma_f32_16x16x32_f16(false, a, false, b, (short)0, acc, false, false);
    }

    const int rowb = 8 * khalf;
    for (int r = 0; r < 8; ++r)
        z[(size_t)(row0 + rowb + r) * K_CB + n] = acc[r];
}

// ---------------- per-row stats: max, logsumexp(z/T), and S += sum(exp(z/eps)) ----------------
__global__ void __launch_bounds__(256) k_rowstats(const float* __restrict__ z,
                                                  float* __restrict__ rowMax,
                                                  float* __restrict__ rowLse,
                                                  float* __restrict__ S) {
    __shared__ float red[256];
    const int b = blockIdx.x, t = threadIdx.x;
    const float* zr = z + (size_t)b * K_CB;

    float v[8];
    float m = -1e30f;
    for (int j = 0; j < 8; ++j) { v[j] = zr[t + j * 256]; m = fmaxf(m, v[j]); }
    red[t] = m; __syncthreads();
    for (int s = 128; s > 0; s >>= 1) { if (t < s) red[t] = fmaxf(red[t], red[t + s]); __syncthreads(); }
    m = red[0];
    __syncthreads();

    float le = 0.f, ee = 0.f;
    for (int j = 0; j < 8; ++j) {
        le += __expf((v[j] - m) * INV_T);   // softmax partition (shifted)
        ee += __expf(v[j] * INV_EPS);       // Sinkhorn E row sum (|z|<=1 so <= e^50, safe in f32)
    }
    red[t] = le; __syncthreads();
    for (int s = 128; s > 0; s >>= 1) { if (t < s) red[t] += red[t + s]; __syncthreads(); }
    le = red[0];
    __syncthreads();
    red[t] = ee; __syncthreads();
    for (int s = 128; s > 0; s >>= 1) { if (t < s) red[t] += red[t + s]; __syncthreads(); }

    if (t == 0) {
        rowMax[b] = m;
        rowLse[b] = __logf(le);
        atomicAdd(&S[b & 1], red[0]);
    }
}

__global__ void k_beta_init(const float* __restrict__ S, float* __restrict__ beta) {
    int b = blockIdx.x * 256 + threadIdx.x;
    if (b < TWO_B) beta[b] = 1.0f / S[b & 1];
}

// ---------------- Sinkhorn column pass: C[p][k] = sum_b E(z[b,k]) * beta[b] ----------------
__global__ void __launch_bounds__(256) k_colsum(const float* __restrict__ z,
                                                const float* __restrict__ beta,
                                                float* __restrict__ C) {
    const int k  = blockIdx.x * 256 + threadIdx.x;
    const int r0 = blockIdx.y * 128;     // always even
    float a0 = 0.f, a1 = 0.f;
    for (int r = 0; r < 128; r += 2) {
        int row = r0 + r;
        a0 += __expf(z[(size_t)row * K_CB + k] * INV_EPS) * beta[row];
        a1 += __expf(z[(size_t)(row + 1) * K_CB + k] * INV_EPS) * beta[row + 1];
    }
    atomicAdd(&C[k], a0);
    atomicAdd(&C[K_CB + k], a1);
}

__global__ void k_alpha(const float* __restrict__ C, float* __restrict__ alpha) {
    int i = blockIdx.x * 256 + threadIdx.x;
    if (i < 2 * K_CB) alpha[i] = 1.0f / ((float)K_CB * C[i]);
}

// ---------------- Sinkhorn row pass: beta[b] = 1 / (B * sum_k E * alpha[p][k]) ----------------
__global__ void __launch_bounds__(256) k_rowbeta(const float* __restrict__ z,
                                                 const float* __restrict__ alpha,
                                                 float* __restrict__ beta) {
    __shared__ float red[256];
    const int b = blockIdx.x, t = threadIdx.x;
    const float* al = alpha + (size_t)(b & 1) * K_CB;
    const float* zr = z + (size_t)b * K_CB;
    float acc = 0.f;
    for (int j = 0; j < 8; ++j) {
        int k = t + j * 256;
        acc += __expf(zr[k] * INV_EPS) * al[k];
    }
    red[t] = acc; __syncthreads();
    for (int s = 128; s > 0; s >>= 1) { if (t < s) red[t] += red[t + s]; __syncthreads(); }
    if (t == 0) beta[b] = 1.0f / ((float)B_HALF * red[0]);
}

// ---------------- final: out[i] = -0.5 * (sum q1*logp2 + sum q2*logp1) ----------------
__global__ void __launch_bounds__(256) k_final(const float* __restrict__ z,
                                               const float* __restrict__ alpha,
                                               const float* __restrict__ beta,
                                               const float* __restrict__ rowMax,
                                               const float* __restrict__ rowLse,
                                               float* __restrict__ out) {
    __shared__ float red[256];
    const int i = blockIdx.x, t = threadIdx.x;
    const int b1 = 2 * i, b2 = 2 * i + 1;
    const float* z1 = z + (size_t)b1 * K_CB;
    const float* z2 = z + (size_t)b2 * K_CB;
    const float be1 = beta[b1] * (float)B_HALF;
    const float be2 = beta[b2] * (float)B_HALF;
    const float m1 = rowMax[b1], l1 = rowLse[b1];
    const float m2 = rowMax[b2], l2 = rowLse[b2];

    float acc = 0.f;
    for (int j = 0; j < 8; ++j) {
        int k = t + j * 256;
        float zv1 = z1[k], zv2 = z2[k];
        float q1  = __expf(zv1 * INV_EPS) * alpha[k] * be1;          // parity 0
        float q2  = __expf(zv2 * INV_EPS) * alpha[K_CB + k] * be2;   // parity 1
        float lp1 = (zv1 - m1) * INV_T - l1;
        float lp2 = (zv2 - m2) * INV_T - l2;
        acc += q1 * lp2 + q2 * lp1;
    }
    red[t] = acc; __syncthreads();
    for (int s = 128; s > 0; s >>= 1) { if (t < s) red[t] += red[t + s]; __syncthreads(); }
    if (t == 0) out[i] = -0.5f * red[0];
}

// ---------------- launcher ----------------
extern "C" void kernel_launch(void* const* d_in, const int* in_sizes, int n_in,
                              void* d_out, int out_size, void* d_ws, size_t ws_size,
                              hipStream_t stream) {
    (void)in_sizes; (void)n_in; (void)out_size; (void)ws_size;
    const float* x      = (const float*)d_in[0];
    const float* head_w = (const float*)d_in[1];
    const float* head_b = (const float*)d_in[2];
    const float* cb_v   = (const float*)d_in[3];
    const float* cb_g   = (const float*)d_in[4];
    float* out = (float*)d_out;

    // workspace carve-up (~145 MB); z kept f32 so exp(z*50) stays accurate & L2-resident
    char* ws = (char*)d_ws;
    size_t off = 0;
    float*    z      = (float*)(ws + off);     off += (size_t)TWO_B * K_CB * 4;   // 128 MB
    _Float16* h16    = (_Float16*)(ws + off);  off += (size_t)TWO_B * CDIM * 2;   // 8 MB
    _Float16* Wh     = (_Float16*)(ws + off);  off += (size_t)CDIM * INP * 2;     // 512 KB
    _Float16* wn     = (_Float16*)(ws + off);  off += (size_t)K_CB * CDIM * 2;    // 1 MB
    float*    rowMax = (float*)(ws + off);     off += (size_t)TWO_B * 4;
    float*    rowLse = (float*)(ws + off);     off += (size_t)TWO_B * 4;
    float*    beta   = (float*)(ws + off);     off += (size_t)TWO_B * 4;
    float*    C      = (float*)(ws + off);     off += (size_t)2 * K_CB * 4;
    float*    alpha  = (float*)(ws + off);     off += (size_t)2 * K_CB * 4;
    float*    S      = (float*)(ws + off);     off += 256;

    (void)hipMemsetAsync(S, 0, 2 * sizeof(float), stream);
    k_cvt_headw<<<(CDIM * INP) / 256, 256, 0, stream>>>(head_w, Wh);
    k_prep_wn<<<K_CB, 256, 0, stream>>>(cb_v, cb_g, wn);

    k_head_gemm<<<TWO_B / 16, 256, 0, stream>>>(x, Wh, head_b, h16);
    k_z_gemm<<<dim3(TWO_B / 16, K_CB / 128), 256, 0, stream>>>(h16, wn, z);

    k_rowstats<<<TWO_B, 256, 0, stream>>>(z, rowMax, rowLse, S);
    k_beta_init<<<TWO_B / 256, 256, 0, stream>>>(S, beta);

    for (int it = 0; it < 3; ++it) {
        (void)hipMemsetAsync(C, 0, 2 * K_CB * sizeof(float), stream);
        k_colsum<<<dim3(K_CB / 256, TWO_B / 128), 256, 0, stream>>>(z, beta, C);
        k_alpha<<<(2 * K_CB + 255) / 256, 256, 0, stream>>>(C, alpha);
        k_rowbeta<<<TWO_B, 256, 0, stream>>>(z, alpha, beta);
    }

    k_final<<<B_HALF, 256, 0, stream>>>(z, alpha, beta, rowMax, rowLse, out);
}
